// SelfAttention_43937515438264
// MI455X (gfx1250) — matrix operand order repaired and checked by reading, not compiled, not run on hardware
//
#include <hip/hip_runtime.h>
#include <math.h>

// ---------------------------------------------------------------------------
// SAGAN self-attention, fused flash-attention style, f16 WMMA on gfx1250.
//   B=4, C=256, Cq=32, H=W=64, N=4096
//   Kernel 0: one-shot weight f32->f16 conversion into d_ws
//   Kernel 1: qT[B][N][32], kT[B][N][32], vh[B][C][N] (f16), channel-split
//             across 2 waves; all fragment loads fenced before the WMMA chain
//             with sched_barrier so they stay batched (progressive waits).
//   Kernel 2: flash attention (S^T = K Q^T, online softmax, O = V P^T),
//             j-step 64, c-half split; K fragments pipelined across
//             iterations, V fragments batch-loaded per chunk, both fenced.
// ---------------------------------------------------------------------------

typedef __attribute__((ext_vector_type(16))) _Float16 v16h;
typedef __attribute__((ext_vector_type(8)))  _Float16 v8h;
typedef __attribute__((ext_vector_type(8)))  float    v8f;

static constexpr int Bb = 4;
static constexpr int Cc = 256;
static constexpr int CQ = 32;
static constexpr int Nn = 4096;   // 64*64

#if defined(__has_builtin)
#if __has_builtin(__builtin_amdgcn_sched_barrier)
#define SCHED_FENCE() __builtin_amdgcn_sched_barrier(0)
#endif
#endif
#ifndef SCHED_FENCE
#define SCHED_FENCE() asm volatile("" ::: "memory")
#endif

__device__ __forceinline__ v8f wmma_f16(v16h a, v16h b, v8f c) {
  return __builtin_amdgcn_wmma_f32_16x16x32_f16(
      /*neg_a=*/false, a, /*neg_b=*/false, b,
      /*c_mod=*/(short)0, c, /*reuse_a=*/false, /*reuse_b=*/false);
}

// ---------------------------------------------------------------------------
// Kernel 0: convert weights to f16 once (weights are L2-resident afterwards).
// ---------------------------------------------------------------------------
__global__ __launch_bounds__(256)
void cvt_weights_kernel(const float* __restrict__ Wq,
                        const float* __restrict__ Wk,
                        const float* __restrict__ Wv,
                        _Float16* __restrict__ Wqh,
                        _Float16* __restrict__ Wkh,
                        _Float16* __restrict__ Wvh)
{
  const int i = blockIdx.x * 256 + threadIdx.x;
  if (i < CQ * Cc) {
    Wqh[i] = (_Float16)Wq[i];
    Wkh[i] = (_Float16)Wk[i];
  }
  if (i < Cc * Cc) {
    Wvh[i] = (_Float16)Wv[i];
  }
}

// ---------------------------------------------------------------------------
// Kernel 1: q/k/v 1x1-conv projections as WMMA GEMMs.
// One wave per (batch, 16-wide n tile, channel half h).
//   wave h computes: q o-tile h, k o-tile h, v o-tiles h*8 .. h*8+7
// ---------------------------------------------------------------------------
__global__ __launch_bounds__(32)
__attribute__((amdgpu_waves_per_eu(2)))
void qkv_proj_kernel(const float* __restrict__ x,
                     const _Float16* __restrict__ Wqh, const float* __restrict__ bq,
                     const _Float16* __restrict__ Wkh, const float* __restrict__ bk,
                     const _Float16* __restrict__ Wvh, const float* __restrict__ bv,
                     _Float16* __restrict__ qT, _Float16* __restrict__ kT,
                     _Float16* __restrict__ vh)
{
  const int lane  = threadIdx.x;
  const int l16   = lane & 15;
  const int hid   = lane >> 4;               // 0 = lanes 0-15, 1 = lanes 16-31
  const int per_b = (Nn / 16) * 2;
  const int b     = blockIdx.x / per_b;
  const int rem   = blockIdx.x % per_b;
  const int n0    = (rem >> 1) * 16;
  const int h     = rem & 1;                 // channel half

  v8f accq = {};
  v8f acck = {};
  v8f accv[8] = {};

  // A-fragment loader: W is [o][c] row-major f16.
  // lane<16: row o0+l16, K chunks c0+{0..7} and c0+{16..23}
  // lane>=16: row o0+l16, K chunks c0+{8..15} and c0+{24..31}
  auto loadW = [&](const _Float16* __restrict__ Wm, int o0, int c0) -> v16h {
    const _Float16* wp = Wm + (size_t)(o0 + l16) * Cc + c0 + hid * 8;
    v8h a0 = *reinterpret_cast<const v8h*>(wp);
    v8h a1 = *reinterpret_cast<const v8h*>(wp + 16);
    v16h a;
#pragma unroll
    for (int i = 0; i < 8; ++i) { a[i] = a0[i]; a[8 + i] = a1[i]; }
    return a;
  };

  // B-fragment: x[b][c][n], lane column n = n0+l16,
  // lane<16: K = c0+{0..7,16..23} ; lane>=16: K = c0+{8..15,24..31} (stride-N gathers + cvt)
  auto loadX = [&](int c0) -> v16h {
    const float* xp = x + ((size_t)b * Cc + c0 + hid * 8) * Nn + n0 + l16;
    v16h xb;
#pragma unroll
    for (int i = 0; i < 16; ++i) xb[i] = (_Float16)xp[(size_t)(i + (i & 8)) * Nn];
    return xb;
  };

  v16h xb = loadX(0);
#pragma unroll 1
  for (int c0 = 0; c0 < Cc; c0 += 32) {
    // issue ALL fragment loads for this K step up front...
    v16h aq = loadW(Wqh, h * 16, c0);
    v16h ak = loadW(Wkh, h * 16, c0);
    v16h av[8];
#pragma unroll
    for (int t = 0; t < 8; ++t) av[t] = loadW(Wvh, h * 128 + t * 16, c0);
    v16h xb_n = loadX((c0 + 32) & (Cc - 1));

    // ...and forbid the scheduler from sinking them into the WMMA chain:
    // all 10 fragments stay live -> distinct registers -> progressive waits.
    SCHED_FENCE();

    accq = wmma_f16(aq, xb, accq);
    acck = wmma_f16(ak, xb, acck);
#pragma unroll
    for (int t = 0; t < 8; ++t) accv[t] = wmma_f16(av[t], xb, accv[t]);

    xb = xb_n;
  }

  // ---- store qT / kT as [b][n][CQ]: D lane holds n = n0+l16 fixed,
  //      o = h*16 + hid*8 + r  -> 8 contiguous halves -> one b128 store.
  {
    v8h oq, ok;
#pragma unroll
    for (int r = 0; r < 8; ++r) {
      oq[r] = (_Float16)(accq[r] + bq[h * 16 + hid * 8 + r]);
      ok[r] = (_Float16)(acck[r] + bk[h * 16 + hid * 8 + r]);
    }
    size_t off = ((size_t)b * Nn + n0 + l16) * CQ + h * 16 + hid * 8;
    *reinterpret_cast<v8h*>(qT + off) = oq;
    *reinterpret_cast<v8h*>(kT + off) = ok;
  }

  // ---- store v as [b][c][n]: transpose each 16x16 D tile through LDS so the
  //      global stores are packed b128 along n.
  __shared__ __align__(64) _Float16 vt[16][16];
#pragma unroll 1
  for (int t = 0; t < 8; ++t) {
    const int o0 = h * 128 + t * 16;
    __syncthreads();
#pragma unroll
    for (int r = 0; r < 8; ++r)
      vt[hid * 8 + r][l16] = (_Float16)(accv[t][r] + bv[o0 + hid * 8 + r]);
    __syncthreads();
    v8h row = *reinterpret_cast<const v8h*>(&vt[l16][hid * 8]);
    *reinterpret_cast<v8h*>(
        vh + ((size_t)b * Cc + o0 + l16) * Nn + n0 + hid * 8) = row;
  }
}

// ---------------------------------------------------------------------------
// Kernel 2: flash attention. One wave per (batch, 16-query i tile, c-half).
//   S^T tiles = K_tile(16j x 32c, A) x Q^T(32c x 16i, B)   [4 WMMA / 64 j]
//   online softmax per column i  (i == lane%16 for both S^T and O layouts)
//   O[c][i] += V_tile(16c x 32j, A) x P^T(32j x 16i, B)    [16 WMMA / 64 j]
// ---------------------------------------------------------------------------
__global__ __launch_bounds__(32)
__attribute__((amdgpu_waves_per_eu(2)))
void flash_attn_kernel(const _Float16* __restrict__ qT,
                       const _Float16* __restrict__ kT,
                       const _Float16* __restrict__ vh,
                       float* __restrict__ out)
{
  const int lane  = threadIdx.x;
  const int l16   = lane & 15;
  const int hid   = lane >> 4;
  const int per_b = (Nn / 16) * 2;
  const int b     = blockIdx.x / per_b;
  const int rem   = blockIdx.x % per_b;
  const int i0    = (rem >> 1) * 16;
  const int cbase = (rem & 1) * (Cc / 2);   // 0 or 128: this wave's 8 c-tiles

  // Q as B-matrix fragment, resident for the whole kernel:
  // lane col i = i0+l16; lane<16: K=c 0..7,16..23, lane>=16: K=c 8..15,24..31.
  v16h qb;
  {
    const _Float16* qp = qT + ((size_t)b * Nn + i0 + l16) * CQ + hid * 8;
    v8h q0 = *reinterpret_cast<const v8h*>(qp);
    v8h q1 = *reinterpret_cast<const v8h*>(qp + 16);
#pragma unroll
    for (int i = 0; i < 8; ++i) { qb[i] = q0[i]; qb[8 + i] = q1[i]; }
  }

  // K A-fragment loader for j tile (j + jt*16).
  auto loadK = [&](int j, int jt) -> v16h {
    const _Float16* kp =
        kT + ((size_t)b * Nn + j + jt * 16 + l16) * CQ + hid * 8;
    v8h a0 = *reinterpret_cast<const v8h*>(kp);
    v8h a1 = *reinterpret_cast<const v8h*>(kp + 16);
    v16h a;
#pragma unroll
    for (int i = 0; i < 8; ++i) { a[i] = a0[i]; a[8 + i] = a1[i]; }
    return a;
  };

  // V A-fragment loader for channel tile t, j chunk (32 wide) at j.
  auto loadV = [&](int t, int j) -> v16h {
    const _Float16* vp =
        vh + ((size_t)b * Cc + cbase + t * 16 + l16) * Nn + j + hid * 8;
    v8h a0 = *reinterpret_cast<const v8h*>(vp);
    v8h a1 = *reinterpret_cast<const v8h*>(vp + 16);
    v16h va;
#pragma unroll
    for (int i = 0; i < 8; ++i) { va[i] = a0[i]; va[8 + i] = a1[i]; }
    return va;
  };

  float m_run = -INFINITY;
  float l_run = 0.0f;
  v8f accO[8] = {};

  __shared__ __align__(64) _Float16 P[16][64];   // [i][j] f16 probabilities

  // preload K fragments for the first iteration.
  v16h ka[4];
#pragma unroll
  for (int jt = 0; jt < 4; ++jt) ka[jt] = loadK(0, jt);

#pragma unroll 1
  for (int j0 = 0; j0 < Nn; j0 += 64) {
    v8f zero = {};
    v8f s[4];
#pragma unroll
    for (int jt = 0; jt < 4; ++jt) s[jt] = wmma_f16(ka[jt], qb, zero);

    // issue next iteration's K fragments now: they get the whole softmax +
    // GEMM2 phase to land (wrapped index keeps the address valid at the end).
    const int jn = (j0 + 64) & (Nn - 1);
    __builtin_prefetch(kT + ((size_t)b * Nn + jn + 64) * CQ, 0, 1);
#pragma unroll
    for (int jt = 0; jt < 4; ++jt) ka[jt] = loadK(jn, jt);
    SCHED_FENCE();   // pin the K loads here, ~20 WMMAs ahead of their use

    // --- online softmax over column i (this lane + partner lane^16).
    float mx = s[0][0];
#pragma unroll
    for (int jt = 0; jt < 4; ++jt)
#pragma unroll
      for (int r = 0; r < 8; ++r) mx = fmaxf(mx, s[jt][r]);
    mx = fmaxf(mx, __shfl_xor(mx, 16, 32));
    const float m_new = fmaxf(m_run, mx);
    const float alpha = __expf(m_run - m_new);

    v8h p[4];
    float ls = 0.0f;
#pragma unroll
    for (int jt = 0; jt < 4; ++jt)
#pragma unroll
      for (int r = 0; r < 8; ++r) {
        float e = __expf(s[jt][r] - m_new);
        ls += e;
        p[jt][r] = (_Float16)e;
      }
    ls += __shfl_xor(ls, 16, 32);
    l_run = l_run * alpha + ls;
    m_run = m_new;

    // rescale O accumulators (per-lane scalar: lane <-> column i).
#pragma unroll
    for (int t = 0; t < 8; ++t)
#pragma unroll
      for (int r = 0; r < 8; ++r) accO[t][r] *= alpha;

    // --- P^T -> LDS [i][j] (packed b128), then read back as B fragments.
    __syncthreads();
#pragma unroll
    for (int jt = 0; jt < 4; ++jt)
      *reinterpret_cast<v8h*>(&P[l16][jt * 16 + hid * 8]) = p[jt];
    __syncthreads();

    // --- O += V_tile x P^T : two K=32 chunks of j, 8 channel tiles each.
    //     All 8 V fragments of a chunk are loaded, fenced, then consumed.
#pragma unroll
    for (int jk = 0; jk < 2; ++jk) {
      v16h va[8];
#pragma unroll
      for (int t = 0; t < 8; ++t) va[t] = loadV(t, j0 + jk * 32);
      SCHED_FENCE();   // keep the 8 V fragments batched ahead of the WMMAs

      v16h pb;
      {
        const v8h* pp =
            reinterpret_cast<const v8h*>(&P[l16][jk * 32 + hid * 8]);
        v8h b0 = pp[0], b1 = pp[2];
#pragma unroll
        for (int i = 0; i < 8; ++i) { pb[i] = b0[i]; pb[8 + i] = b1[i]; }
      }
#pragma unroll
      for (int t = 0; t < 8; ++t) accO[t] = wmma_f16(va[t], pb, accO[t]);
    }
  }

  // --- finalize: divide by softmax denominator, store out[b][c][i] f32.
  const float inv = 1.0f / l_run;
#pragma unroll
  for (int t = 0; t < 8; ++t) {
#pragma unroll
    for (int r = 0; r < 8; ++r) {
      out[((size_t)b * Cc + cbase + t * 16 + hid * 8 + r) * Nn + i0 + l16] =
          accO[t][r] * inv;
    }
  }
}

// ---------------------------------------------------------------------------
extern "C" void kernel_launch(void* const* d_in, const int* in_sizes, int n_in,
                              void* d_out, int out_size, void* d_ws,
                              size_t ws_size, hipStream_t stream)
{
  const float* x  = (const float*)d_in[0];
  const float* Wq = (const float*)d_in[1];
  const float* bq = (const float*)d_in[2];
  const float* Wk = (const float*)d_in[3];
  const float* bk = (const float*)d_in[4];
  const float* Wv = (const float*)d_in[5];
  const float* bv = (const float*)d_in[6];
  float* out = (float*)d_out;

  char* ws = (char*)d_ws;
  const size_t qkBytes = (size_t)Bb * Nn * CQ * sizeof(_Float16);   // 1 MB
  const size_t vBytes  = (size_t)Bb * Cc * Nn * sizeof(_Float16);   // 8 MB
  const size_t wqBytes = (size_t)CQ * Cc * sizeof(_Float16);        // 16 KB
  _Float16* qT  = (_Float16*)(ws);
  _Float16* kT  = (_Float16*)(ws + qkBytes);
  _Float16* vh  = (_Float16*)(ws + 2 * qkBytes);
  _Float16* Wqh = (_Float16*)(ws + 2 * qkBytes + vBytes);
  _Float16* Wkh = (_Float16*)(ws + 2 * qkBytes + vBytes + wqBytes);
  _Float16* Wvh = (_Float16*)(ws + 2 * qkBytes + vBytes + 2 * wqBytes);

  cvt_weights_kernel<<<(Cc * Cc + 255) / 256, 256, 0, stream>>>(
      Wq, Wk, Wv, Wqh, Wkh, Wvh);

  const int blocks1 = Bb * (Nn / 16) * 2;    // 2048 waves (channel split)
  qkv_proj_kernel<<<blocks1, 32, 0, stream>>>(x, Wqh, bq, Wkh, bk, Wvh, bv,
                                              qT, kT, vh);

  const int blocks2 = Bb * (Nn / 16) * 2;    // 2048 waves (c split in half)
  flash_attn_kernel<<<blocks2, 32, 0, stream>>>(qT, kT, vh, out);
}
